// GlobalSparseAttetionModule_23579370455478
// MI455X (gfx1250) — matrix operand order repaired and checked by reading, not compiled, not run on hardware
//
#include <hip/hip_runtime.h>
#include <math.h>

// ---------------------------------------------------------------------------
// Types for CDNA5 WMMA (wave32): D(16x16 f32) = A(16x32 bf16) x B(32x16 bf16) + C
// ---------------------------------------------------------------------------
typedef __bf16 bf16;
typedef __attribute__((ext_vector_type(16))) __bf16 v16bf;
typedef __attribute__((ext_vector_type(8)))  float  v8f;
typedef __attribute__((ext_vector_type(4)))  unsigned int u32x4;
typedef __attribute__((ext_vector_type(8)))  int i32x8;
typedef __attribute__((ext_vector_type(4)))  int i32x4;

#define C_DIM   256
#define HEADS   8
#define HD      32            // head dim
#define N1      1024          // 32*32 tokens after ::2 downsample
#define N2      4096          // 64*64 tokens after upsample
#define BATCH   8
#define SCALE   0.17677669529663689f   // 32^-0.5
#define EPS_LN  1e-5f

// workspace layout (bytes)
#define QKV_ELEMS   (BATCH * HEADS * N1 * HD)          // per tensor, bf16
#define OFF_Q       ((size_t)0)
#define OFF_K       ((size_t)QKV_ELEMS * 2)
#define OFF_V       ((size_t)QKV_ELEMS * 4)            // stored transposed: [b][h][d][n]
#define OFF_O       ((size_t)QKV_ELEMS * 6)            // fp32 [B][C][N1]  (8 MB)
#define OFF_UP      (OFF_O + (size_t)BATCH * C_DIM * N1 * 4)   // fp32 [B][C][N2] (33.5 MB)
#define OFF_ST      (OFF_UP + (size_t)BATCH * C_DIM * N2 * 4)  // fp32 [B][2]

// LDS tile strides (elements) — conflict-free fragment reads; TDM pad fields
// reproduce these strides in hardware (pad 2dw/16dw -> 36, pad 4dw/64dw -> 136).
#define TSTR  36
#define PSTR  136

// ---------------------------------------------------------------------------
// Tensor Data Mover: build D# per ISA 08_async_tensor §8.3/8.4 and issue.
// group0: {count=1, lds_addr, global_addr lo, global_addr hi | type=2}
// group1 dwords passed in (wg_mask=0, data_size=2B, pad cfg, dims, strides).
// ---------------------------------------------------------------------------
__device__ __forceinline__ void tdm_load(unsigned lds_off, const void* gp,
                                         unsigned d0, unsigned d2, unsigned d3,
                                         unsigned d4, unsigned d5) {
    const unsigned long long ga = (unsigned long long)(uintptr_t)gp;
    u32x4 g0 = { 1u, lds_off, (unsigned)ga,
                 (unsigned)((ga >> 32) & 0x1FFFFFFull) | 0x80000000u };
    i32x8 g1 = { (int)d0, 0, (int)d2, (int)d3, (int)d4, (int)d5, 0, 0 };
    i32x4 z4 = { 0, 0, 0, 0 };
#if __has_include(<hip/amd_detail/amd_gfx1250_TDM.h>)
    i32x8 z8 = { 0, 0, 0, 0, 0, 0, 0, 0 };
    __builtin_amdgcn_tensor_load_to_lds(g0, g1, z4, z4, z8, 0);
#else
    __builtin_amdgcn_tensor_load_to_lds(g0, g1, z4, z4, 0);
#endif
}
// 1D 4096-elem bf16 copy, LDS pad 2 dwords every 16 dwords (row pitch 36 elems):
//  d0: data_size=1<<16 | pad_en 1<<20 | pad_int 3<<22 | pad_amt 1<<25
#define TDM1D_D0 0x02D10000u
#define TDM1D_D2 0x00010010u   // tensor_dim0 hi=0x10 (1M), tensor_dim1=1
#define TDM1D_D3 0x10000000u   // tile_dim0 = 4096
#define TDM1D_D4 1u            // tile_dim1 = 1
#define TDM1D_D5 0x00100000u   // tensor_dim0_stride = 1M
// 2D 128x32 bf16 tile, global row stride 1024 elems, pad 4 dwords every 64
// dwords (row pitch 136 elems):
#define TDM2D_D0 0x07510000u   // data_size | pad_en | pad_int 5<<22 | pad_amt 3<<25
#define TDM2D_D2 0x00400010u   // tensor_dim0 hi=0x10, tensor_dim1=64
#define TDM2D_D3 0x00800000u   // tile_dim0 = 128
#define TDM2D_D4 32u           // tile_dim1 = 32
#define TDM2D_D5 0x00000400u   // tensor_dim0_stride = 1024

// ---------------------------------------------------------------------------
// Fragment loaders (per ISA 7.12.2 VGPR layouts, wave32)
// ---------------------------------------------------------------------------
__device__ __forceinline__ v16bf load_a_frag(const bf16* lds, int row_base,
                                             int col_base, int stride) {
    const int lane = threadIdx.x & 31;
    const int lo = lane & 15, hi = lane >> 4;
    const bf16* p = lds + (row_base + lo) * stride + col_base + hi * 8;
    v16bf a;
#pragma unroll
    for (int t = 0; t < 8; ++t) { a[t] = p[t]; a[8 + t] = p[16 + t]; }
    return a;
}

// B tile stored transposed in LDS: [n_col][k]; lane half h reads k = 8h+t and 16+8h+t.
__device__ __forceinline__ v16bf load_b_frag(const bf16* lds, int col_base,
                                             int k_base, int stride) {
    const int lane = threadIdx.x & 31;
    const int lo = lane & 15, hi = lane >> 4;
    const bf16* p = lds + (col_base + lo) * stride + k_base + hi * 8;
    v16bf b;
#pragma unroll
    for (int t = 0; t < 8; ++t) { b[t] = p[t]; b[8 + t] = p[16 + t]; }
    return b;
}

__device__ __forceinline__ v8f wmma_bf16(v16bf a, v16bf b, v8f c) {
    return __builtin_amdgcn_wmma_f32_16x16x32_bf16(false, a, false, b,
                                                   (short)0, c, false, false);
}

__device__ __forceinline__ float redmax16(float v) {
#pragma unroll
    for (int m = 8; m >= 1; m >>= 1) v = fmaxf(v, __shfl_xor(v, m, 16));
    return v;
}
__device__ __forceinline__ float redsum16(float v) {
#pragma unroll
    for (int m = 8; m >= 1; m >>= 1) v += __shfl_xor(v, m, 16);
    return v;
}

// ---------------------------------------------------------------------------
// Kernel 0: zero LN stats
// ---------------------------------------------------------------------------
__global__ void k_init_stats(float* stats) {
    if (threadIdx.x < 2 * BATCH) stats[threadIdx.x] = 0.0f;
}

// ---------------------------------------------------------------------------
// Kernel 1: qkv = w_qkv(768x256) @ xs(256x1024) per batch, bf16 WMMA.
// Epilogue scatters Q,K -> [b][h][n][d] (SCALE folded into Q) and V
// pre-transposed -> [b][h][d][n] so attention's PV B-tiles need no transpose.
// ---------------------------------------------------------------------------
__global__ __launch_bounds__(256) void k_qkv_gemm(const float* __restrict__ x,
                                                  const float* __restrict__ w_qkv,
                                                  bf16* __restrict__ qkv) {
    __shared__ __align__(16) bf16 At[128 * TSTR];   // [m_row][k]
    __shared__ __align__(16) bf16 Bt[128 * TSTR];   // [n_col][k] (transposed)

    const int tid = threadIdx.x;
    const int n0 = blockIdx.x * 128;
    const int o0 = blockIdx.y * 128;
    const int b  = blockIdx.z;
    const int w  = tid >> 5;
    const int wm = w >> 2, wn = w & 3;              // 2x4 wave grid -> 64x32 per wave
    const int lane = tid & 31, lo = lane & 15, hi = lane >> 4;

    v8f acc[4][2];
#pragma unroll
    for (int i = 0; i < 4; ++i)
#pragma unroll
        for (int j = 0; j < 2; ++j) acc[i][j] = {};

#pragma unroll 1
    for (int kk = 0; kk < 8; ++kk) {                // K = 256 in steps of 32
        // A: w_qkv rows o0..o0+127, cols kk*32..+31
        {
            const int r = tid >> 1, c0 = (tid & 1) * 16;
            const float* src = w_qkv + (size_t)(o0 + r) * C_DIM + kk * 32 + c0;
#pragma unroll
            for (int i = 0; i < 16; ++i) At[r * TSTR + c0 + i] = (bf16)src[i];
            if (kk < 7) __builtin_prefetch(src + 32, 0, 0);
        }
        // B: xs[b][c][n] = x[b][c][2*hh][2*ww], store transposed [n][c]
        {
            const int c = tid >> 3, nb = (tid & 7) * 16;
            const int cg = kk * 32 + c;
            const float* xb = x + ((size_t)(b * C_DIM + cg)) * 4096;
#pragma unroll
            for (int i = 0; i < 16; ++i) {
                const int n = n0 + nb + i;
                const float v = xb[((n >> 5) * 2) * 64 + (n & 31) * 2];
                Bt[(nb + i) * TSTR + c] = (bf16)v;
            }
            if (kk < 7) __builtin_prefetch(xb + 32 * 4096, 0, 0);
        }
        __syncthreads();

        v16bf af[4], bfg[2];
#pragma unroll
        for (int ms = 0; ms < 4; ++ms) af[ms] = load_a_frag(At, wm * 64 + ms * 16, 0, TSTR);
#pragma unroll
        for (int ns = 0; ns < 2; ++ns) bfg[ns] = load_b_frag(Bt, wn * 32 + ns * 16, 0, TSTR);
#pragma unroll
        for (int ms = 0; ms < 4; ++ms)
#pragma unroll
            for (int ns = 0; ns < 2; ++ns) acc[ms][ns] = wmma_bf16(af[ms], bfg[ns], acc[ms][ns]);
        __syncthreads();
    }

    // epilogue: head/sel/d are constant per fragment (8-aligned base), so
    // resolve them once per (ms) instead of per element.
    const int bh_base = b * HEADS;
#pragma unroll
    for (int ms = 0; ms < 4; ++ms) {
        const int obase = o0 + wm * 64 + ms * 16 + hi * 8;
        const int h = obase / 96;
        const int r = obase - h * 96;
        const int sel = r >> 5;
        const int dbase = r & 31;
#pragma unroll
        for (int ns = 0; ns < 2; ++ns) {
            const int n = n0 + wn * 32 + ns * 16 + lo;
            if (sel == 2) {            // V transposed: [b][h][d][n]
                bf16* dst = qkv + OFF_V / 2 +
                            (((size_t)(bh_base + h) * HD + dbase) << 10) + n;
#pragma unroll
                for (int e = 0; e < 8; ++e) dst[(size_t)e << 10] = (bf16)acc[ms][ns][e];
            } else {                   // Q (scaled) / K: [b][h][n][d], 8 contiguous
                const float scl = (sel == 0) ? SCALE : 1.0f;
                bf16* dst = qkv + (size_t)sel * QKV_ELEMS +
                            (((size_t)(bh_base + h) * N1 + n) << 5) + dbase;
#pragma unroll
                for (int e = 0; e < 8; ++e) dst[e] = (bf16)(acc[ms][ns][e] * scl);
            }
        }
    }
}

// ---------------------------------------------------------------------------
// Kernel 2: flash attention per (b,h), 128-row i-tiles, online softmax.
// Q/K/V tiles staged by the Tensor Data Mover (with hardware LDS padding),
// issued once per workgroup from wave 0, synced via TENSORcnt + barrier.
// ---------------------------------------------------------------------------
__global__ __launch_bounds__(256) void k_attention(const bf16* __restrict__ qkv,
                                                   float* __restrict__ O) {
    __shared__ __align__(16) bf16 smem[128 * TSTR * 2 + 32 * PSTR + 8 * 16 * PSTR];
    bf16* Qt = smem;                        // [128][TSTR]  (i rows, d)
    bf16* Kt = Qt + 128 * TSTR;             // [128][TSTR]  (j rows, d)  -> B for QK^T
    bf16* Vt = Kt + 128 * TSTR;             // [32][PSTR]   (d, j)       -> B for PV
    bf16* Pt = Vt + 32 * PSTR;              // [8 waves][16][PSTR]

    const int tid = threadIdx.x;
    const int i0 = blockIdx.x * 128;
    const int bh = blockIdx.y;
    const int b = bh >> 3, h = bh & 7;
    const int w = tid >> 5, lane = tid & 31, lo = lane & 15, hi = lane >> 4;

    const bf16* Qg = qkv + OFF_Q / 2 + ((size_t)bh << 15);          // [n][d]
    const bf16* Kg = qkv + OFF_K / 2 + ((size_t)bh << 15);          // [n][d]
    const bf16* Vg = qkv + OFF_V / 2 + ((size_t)bh * HD << 10);     // [d][n]

    if (tid < 32) {      // wave-uniform: TDM ignores EXEC, issue exactly once
        tdm_load((unsigned)(uintptr_t)Qt, Qg + ((size_t)i0 << 5),
                 TDM1D_D0, TDM1D_D2, TDM1D_D3, TDM1D_D4, TDM1D_D5);
        __builtin_amdgcn_s_wait_tensorcnt(0);
    }
    __syncthreads();
    const v16bf qf = load_a_frag(Qt, w * 16, 0, TSTR);   // 16 rows x 32 d, in regs

    float mrow[8], lrow[8];
#pragma unroll
    for (int v = 0; v < 8; ++v) { mrow[v] = -INFINITY; lrow[v] = 0.0f; }
    v8f oacc[2] = {{}, {}};
    bf16* pw = Pt + w * 16 * PSTR;

#pragma unroll 1
    for (int jt = 0; jt < 8; ++jt) {
        __syncthreads();             // prior tile fully consumed
        if (tid < 32) {
            tdm_load((unsigned)(uintptr_t)Kt, Kg + ((size_t)(jt * 128) << 5),
                     TDM1D_D0, TDM1D_D2, TDM1D_D3, TDM1D_D4, TDM1D_D5);
            tdm_load((unsigned)(uintptr_t)Vt, Vg + jt * 128,
                     TDM2D_D0, TDM2D_D2, TDM2D_D3, TDM2D_D4, TDM2D_D5);
            __builtin_amdgcn_s_wait_tensorcnt(0);
        }
        __syncthreads();             // TDM-written LDS visible to all waves

        // S strip = Q(16x32) @ K^T  -> 16 x 128 in 8 C-fragments
        v8f sf[8];
#pragma unroll
        for (int nb = 0; nb < 8; ++nb)
            sf[nb] = wmma_bf16(qf, load_b_frag(Kt, nb * 16, 0, TSTR), (v8f){});

        // online softmax per row (rows live in lane halves, elems 0..7)
        float mnew[8], alpha[8];
#pragma unroll
        for (int v = 0; v < 8; ++v) {
            float mx = -INFINITY;
#pragma unroll
            for (int nb = 0; nb < 8; ++nb) mx = fmaxf(mx, sf[nb][v]);
            mx = redmax16(mx);
            const float mn = fmaxf(mrow[v], mx);
            alpha[v] = __expf(mrow[v] - mn);
            mnew[v] = mn; mrow[v] = mn;
        }
#pragma unroll
        for (int v = 0; v < 8; ++v) {
            float rs = 0.0f;
#pragma unroll
            for (int nb = 0; nb < 8; ++nb) {
                const float p = __expf(sf[nb][v] - mnew[v]);
                sf[nb][v] = p; rs += p;
            }
            lrow[v] = lrow[v] * alpha[v] + redsum16(rs);
        }
#pragma unroll
        for (int ds = 0; ds < 2; ++ds)
#pragma unroll
            for (int v = 0; v < 8; ++v) oacc[ds][v] *= alpha[v];

        // spill P (C-layout) to per-wave LDS, reload as A-fragments
#pragma unroll
        for (int nb = 0; nb < 8; ++nb)
#pragma unroll
            for (int v = 0; v < 8; ++v)
                pw[(v + 8 * hi) * PSTR + nb * 16 + lo] = (bf16)sf[nb][v];

#pragma unroll
        for (int kk = 0; kk < 4; ++kk) {
            const v16bf pa = load_a_frag(pw, 0, kk * 32, PSTR);
#pragma unroll
            for (int ds = 0; ds < 2; ++ds)
                oacc[ds] = wmma_bf16(pa, load_b_frag(Vt, ds * 16, kk * 32, PSTR), oacc[ds]);
        }
    }

    // normalize + write O[b][c][i]
#pragma unroll
    for (int ds = 0; ds < 2; ++ds) {
        const int c = h * 32 + ds * 16 + lo;
#pragma unroll
        for (int v = 0; v < 8; ++v) {
            const int i = i0 + w * 16 + v + 8 * hi;
            O[(((size_t)(b * C_DIM + c)) << 10) + i] = oacc[ds][v] / lrow[v];
        }
    }
}

// ---------------------------------------------------------------------------
// Kernel 3: 2x2 per-channel upsample + bias, accumulate batch sum/sumsq.
// ---------------------------------------------------------------------------
__global__ __launch_bounds__(256) void k_upsample_stats(const float* __restrict__ O,
                                                        const float* __restrict__ w_lp,
                                                        const float* __restrict__ b_lp,
                                                        float* __restrict__ up,
                                                        float* __restrict__ stats) {
    const int bc = blockIdx.x;
    const int b = bc >> 8, c = bc & 255;
    const float w00 = w_lp[c * 4 + 0], w01 = w_lp[c * 4 + 1];
    const float w10 = w_lp[c * 4 + 2], w11 = w_lp[c * 4 + 3];
    const float bl = b_lp[c];
    const float* Ob = O + ((size_t)bc << 10);
    float* upb = up + ((size_t)bc << 12);

    float s = 0.0f, ss = 0.0f;
#pragma unroll
    for (int it = 0; it < 4; ++it) {
        const int n = threadIdx.x + it * 256;
        const float o = Ob[n];
        const int hh = n >> 5, ww = n & 31;
        const float v00 = fmaf(o, w00, bl), v01 = fmaf(o, w01, bl);
        const float v10 = fmaf(o, w10, bl), v11 = fmaf(o, w11, bl);
        const int base = (hh * 2) * 64 + ww * 2;
        upb[base] = v00; upb[base + 1] = v01;
        upb[base + 64] = v10; upb[base + 65] = v11;
        s += v00 + v01 + v10 + v11;
        ss += v00 * v00 + v01 * v01 + v10 * v10 + v11 * v11;
    }
#pragma unroll
    for (int m = 16; m >= 1; m >>= 1) { s += __shfl_xor(s, m, 32); ss += __shfl_xor(ss, m, 32); }
    __shared__ float sh[2];
    if (threadIdx.x == 0) { sh[0] = 0.0f; sh[1] = 0.0f; }
    __syncthreads();
    if ((threadIdx.x & 31) == 0) { atomicAdd(&sh[0], s); atomicAdd(&sh[1], ss); }
    __syncthreads();
    if (threadIdx.x == 0) { atomicAdd(&stats[b * 2], sh[0]); atomicAdd(&stats[b * 2 + 1], sh[1]); }
}

// ---------------------------------------------------------------------------
// Kernel 4: LayerNorm-affine fused into proj GEMM: y = w_proj @ normed.
// ---------------------------------------------------------------------------
__global__ __launch_bounds__(256) void k_norm_proj(const float* __restrict__ up,
                                                   const float* __restrict__ w_proj,
                                                   const float* __restrict__ gamma,
                                                   const float* __restrict__ beta,
                                                   const float* __restrict__ stats,
                                                   float* __restrict__ y) {
    __shared__ __align__(16) bf16 At[128 * TSTR];
    __shared__ __align__(16) bf16 Bt[128 * TSTR];

    const int tid = threadIdx.x;
    const int n0 = blockIdx.x * 128;
    const int o0 = blockIdx.y * 128;
    const int b  = blockIdx.z;
    const int w  = tid >> 5, wm = w >> 2, wn = w & 3;
    const int lane = tid & 31, lo = lane & 15, hi = lane >> 4;

    const float cnt = (float)(C_DIM * N2);
    const float mu = stats[b * 2] / cnt;
    const float var = stats[b * 2 + 1] / cnt - mu * mu;
    const float rsig = rsqrtf(var + EPS_LN);

    v8f acc[4][2];
#pragma unroll
    for (int i = 0; i < 4; ++i)
#pragma unroll
        for (int j = 0; j < 2; ++j) acc[i][j] = {};

#pragma unroll 1
    for (int kk = 0; kk < 8; ++kk) {
        {
            const int r = tid >> 1, c0 = (tid & 1) * 16;
            const float* src = w_proj + (size_t)(o0 + r) * C_DIM + kk * 32 + c0;
#pragma unroll
            for (int i = 0; i < 16; ++i) At[r * TSTR + c0 + i] = (bf16)src[i];
            if (kk < 7) __builtin_prefetch(src + 32, 0, 0);
        }
        {
            const int c = tid >> 3, nb = (tid & 7) * 16;
            const int cg = kk * 32 + c;
            const float g = gamma[cg] * rsig, be = beta[cg] - mu * g;
            const float* src = up + ((size_t)(b * C_DIM + cg) << 12) + n0 + nb;
#pragma unroll
            for (int i = 0; i < 16; ++i)
                Bt[(nb + i) * TSTR + c] = (bf16)fmaf(src[i], g, be);
            if (kk < 7) __builtin_prefetch(src + (32ull << 12), 0, 0);
        }
        __syncthreads();

        v16bf af[4], bfg[2];
#pragma unroll
        for (int ms = 0; ms < 4; ++ms) af[ms] = load_a_frag(At, wm * 64 + ms * 16, 0, TSTR);
#pragma unroll
        for (int ns = 0; ns < 2; ++ns) bfg[ns] = load_b_frag(Bt, wn * 32 + ns * 16, 0, TSTR);
#pragma unroll
        for (int ms = 0; ms < 4; ++ms)
#pragma unroll
            for (int ns = 0; ns < 2; ++ns) acc[ms][ns] = wmma_bf16(af[ms], bfg[ns], acc[ms][ns]);
        __syncthreads();
    }

#pragma unroll
    for (int ms = 0; ms < 4; ++ms)
#pragma unroll
        for (int ns = 0; ns < 2; ++ns) {
            const int n = n0 + wn * 32 + ns * 16 + lo;
#pragma unroll
            for (int e = 0; e < 8; ++e) {
                const int o = o0 + wm * 64 + ms * 16 + e + hi * 8;
                y[((size_t)(b * C_DIM + o) << 12) + n] = acc[ms][ns][e];
            }
        }
}

// ---------------------------------------------------------------------------
extern "C" void kernel_launch(void* const* d_in, const int* in_sizes, int n_in,
                              void* d_out, int out_size, void* d_ws, size_t ws_size,
                              hipStream_t stream) {
    const float* x      = (const float*)d_in[0];
    const float* w_qkv  = (const float*)d_in[1];
    const float* w_lp   = (const float*)d_in[2];
    const float* b_lp   = (const float*)d_in[3];
    const float* gamma  = (const float*)d_in[4];
    const float* beta   = (const float*)d_in[5];
    const float* w_proj = (const float*)d_in[6];
    float* y = (float*)d_out;

    char* ws = (char*)d_ws;
    bf16*  qkvb  = (bf16*)ws;                 // Q,K,V bf16 (12.6 MB)
    float* Obuf  = (float*)(ws + OFF_O);      // attn out fp32 (8 MB)
    float* up    = (float*)(ws + OFF_UP);     // upsampled fp32 (33.5 MB)
    float* stats = (float*)(ws + OFF_ST);     // per-batch sum/sumsq

    k_init_stats<<<1, 32, 0, stream>>>(stats);
    k_qkv_gemm<<<dim3(8, 6, BATCH), 256, 0, stream>>>(x, w_qkv, qkvb);
    k_attention<<<dim3(8, BATCH * HEADS), 256, 0, stream>>>(qkvb, Obuf);
    k_upsample_stats<<<BATCH * C_DIM, 256, 0, stream>>>(Obuf, w_lp, b_lp, up, stats);
    k_norm_proj<<<dim3(32, 2, BATCH), 256, 0, stream>>>(up, w_proj, gamma, beta, stats, y);
}
